// ATT_SYN_77163382440827
// MI455X (gfx1250) — matrix operand order repaired and checked by reading, not compiled, not run on hardware
//
#include <hip/hip_runtime.h>
#include <hip/hip_bf16.h>
#include <hip/hip_fp16.h>

// ---------------------------------------------------------------------------
// ATT_SYN for MI455X (gfx1250, wave32, WMMA).
// Dense linear algebra on V_WMMA_F32_16X16X32_F16 (f32 accumulate);
// LSTM xW timestep tiles staged via async global->LDS when available.
// ---------------------------------------------------------------------------

static constexpr int Bc = 8, Tc = 512, Sc = 32, Jc = 64, Dc = 400;
static constexpr int Rc = 256, MDUc = 100, TAGSc = 7;

typedef __attribute__((ext_vector_type(4)))  int      v4i;
typedef __attribute__((address_space(3)))    v4i      lds_v4i;
typedef __attribute__((ext_vector_type(4)))  float    v4f;
typedef __attribute__((ext_vector_type(4)))  _Float16 v4h;
typedef __attribute__((ext_vector_type(8)))  _Float16 v8h;
typedef __attribute__((ext_vector_type(16))) _Float16 v16h;
typedef __attribute__((ext_vector_type(8)))  float    v8f;

#define CAT16(a, b) __builtin_shufflevector((a), (b), 0,1,2,3,4,5,6,7,8,9,10,11,12,13,14,15)
#define WMMA_F16(af, bf, cc) \
  __builtin_amdgcn_wmma_f32_16x16x32_f16(false, (af), false, (bf), (short)0, (cc), false, false)

#ifndef __has_builtin
#define __has_builtin(x) 0
#endif
#if __has_builtin(__builtin_amdgcn_global_load_async_to_lds_b128)
#define USE_ASYNC_LDS 1
#else
#define USE_ASYNC_LDS 0
#endif

#if USE_ASYNC_LDS
__device__ __forceinline__ void async_wait_all() {
#if __has_builtin(__builtin_amdgcn_s_wait_asynccnt)
  __builtin_amdgcn_s_wait_asynccnt(0);
#else
  asm volatile("s_wait_asynccnt 0" ::: "memory");
#endif
}
#endif

__device__ __forceinline__ float sigmf_(float x) { return 1.0f / (1.0f + __expf(-x)); }

__device__ __forceinline__ v8f v8f_zero() {
  v8f z;
#pragma unroll
  for (int i = 0; i < 8; ++i) z[i] = 0.0f;
  return z;
}

// ---------------------------------------------------------------------------
// Tiled WMMA GEMM:  C[M,N] = A[M,K] (f32) @ Bw[N,K]^T (f32, torch weight) + bias
// Block: 256 threads (8 waves), tile 64(M) x 128(N), K step 32, f16 LDS staging.
// Interior tiles: unguarded float4 loads (global_load_b128) + v4h LDS stores.
// ---------------------------------------------------------------------------
__global__ __launch_bounds__(256) void gemm_wmma_nt(
    const float* __restrict__ A, const float* __restrict__ Bw,
    const float* __restrict__ bias, void* __restrict__ C,
    int M, int N, int K, int out_f16) {
  __shared__ _Float16 As[64][32];   // [m][k]
  __shared__ _Float16 Bs[128][32];  // [n][k]  (N-major => contiguous K per lane)

  const int tid = threadIdx.x;
  const int wave = tid >> 5, lane = tid & 31;
  const int hsel = lane >> 4, l16 = lane & 15;
  const int wm = wave >> 1, wn = wave & 1;
  const int m0 = blockIdx.y * 64;
  const int n0 = blockIdx.x * 128;
  const bool intM = (m0 + 64 <= M);
  const bool intN = (n0 + 128 <= N);

  v8f acc[4];
#pragma unroll
  for (int i = 0; i < 4; ++i) acc[i] = v8f_zero();

  for (int k0 = 0; k0 < K; k0 += 32) {
    const bool kfull = (k0 + 32 <= K);
    __syncthreads();
    // stage A 64x32 (512 float4, 2 per thread)
#pragma unroll
    for (int j = 0; j < 2; ++j) {
      int idx = tid + 256 * j;
      int r = idx >> 3, kq = (idx & 7) << 2;
      int gm = m0 + r, gk = k0 + kq;
      v4f v;
      if (intM && kfull) {
        v = *(const v4f*)&A[(size_t)gm * K + gk];
      } else {
#pragma unroll
        for (int e = 0; e < 4; ++e)
          v[e] = (gm < M && (gk + e) < K) ? A[(size_t)gm * K + gk + e] : 0.0f;
      }
      v4h h;
#pragma unroll
      for (int e = 0; e < 4; ++e) h[e] = (_Float16)v[e];
      *(v4h*)&As[r][kq] = h;
    }
    // stage B 128x32 (1024 float4, 4 per thread)
#pragma unroll
    for (int j = 0; j < 4; ++j) {
      int idx = tid + 256 * j;
      int r = idx >> 3, kq = (idx & 7) << 2;
      int gn = n0 + r, gk = k0 + kq;
      v4f v;
      if (intN && kfull) {
        v = *(const v4f*)&Bw[(size_t)gn * K + gk];
      } else {
#pragma unroll
        for (int e = 0; e < 4; ++e)
          v[e] = (gn < N && (gk + e) < K) ? Bw[(size_t)gn * K + gk + e] : 0.0f;
      }
      v4h h;
#pragma unroll
      for (int e = 0; e < 4; ++e) h[e] = (_Float16)v[e];
      *(v4h*)&Bs[r][kq] = h;
    }
    __syncthreads();

    // A fragment: lanes 0-15 -> K {0..7,16..23}; lanes 16-31 -> K {8..15,24..31}
    v8h a0 = *(const v8h*)&As[wm * 16 + l16][hsel * 8];
    v8h a1 = *(const v8h*)&As[wm * 16 + l16][hsel * 8 + 16];
    v16h af = CAT16(a0, a1);
#pragma unroll
    for (int nt = 0; nt < 4; ++nt) {
      int bn = wn * 64 + nt * 16 + l16;  // N index (same for both lane halves)
      v8h b0 = *(const v8h*)&Bs[bn][hsel * 8];
      v8h b1 = *(const v8h*)&Bs[bn][hsel * 8 + 16];
      v16h bf = CAT16(b0, b1);
      acc[nt] = WMMA_F16(af, bf, acc[nt]);
    }
  }

  if (intM && intN) {
#pragma unroll
    for (int nt = 0; nt < 4; ++nt) {
      int col = n0 + wn * 64 + nt * 16 + l16;
      float bv = bias ? bias[col] : 0.0f;
#pragma unroll
      for (int v = 0; v < 8; ++v) {
        int row = m0 + wm * 16 + v + hsel * 8;
        float val = acc[nt][v] + bv;
        if (out_f16) ((_Float16*)C)[(size_t)row * N + col] = (_Float16)val;
        else         ((float*)C)[(size_t)row * N + col] = val;
      }
    }
  } else {
#pragma unroll
    for (int nt = 0; nt < 4; ++nt) {
      int col = n0 + wn * 64 + nt * 16 + l16;
      if (col >= N) continue;
      float bv = bias ? bias[col] : 0.0f;
#pragma unroll
      for (int v = 0; v < 8; ++v) {
        int row = m0 + wm * 16 + v + hsel * 8;
        if (row < M) {
          float val = acc[nt][v] + bv;
          if (out_f16) ((_Float16*)C)[(size_t)row * N + col] = (_Float16)val;
          else         ((float*)C)[(size_t)row * N + col] = val;
        }
      }
    }
  }
}

// ---------------------------------------------------------------------------
// Persistent LSTM scan. One block = 16 batch rows (padded). 1024 thr = 32 waves.
// Per step:
//   phase0: stage xW[t] (f16, biases folded) tile into gL via async->LDS
//   WMMA:   h @ Whh^T (wave w owns gate cols [32w,32w+32))
//   phase3: gL += accumulators (LDS RMW transpose)
//   phase4: fused sigmoid/tanh state update; h kept in LDS (f16)
// WhhT is f16 [1024,256] (torch Whh layout == B^T layout we need), L2-resident.
// ---------------------------------------------------------------------------
__global__ __launch_bounds__(1024) void lstm_scan_wmma(
    const _Float16* __restrict__ xW, const _Float16* __restrict__ WhhT,
    float* __restrict__ hs, float* __restrict__ cs,
    int nRows, int T, int in_rev, int out_rev) {
  __shared__ _Float16 hL[16][256];    // current hidden (A matrix), 8 KB
  __shared__ _Float16 gL[16][1024];   // gate pre-activations, 32 KB

  const int tid = threadIdx.x;
  const int wave = tid >> 5, lane = tid & 31;
  const int hsel = lane >> 4, l16 = lane & 15;
  const int row0 = blockIdx.x * 16;
  const int ncol0 = wave * 32;

#pragma unroll
  for (int j = 0; j < 4; ++j) {  // zero-init h
    int idx = tid + 1024 * j;
    hL[idx >> 8][idx & 255] = (_Float16)0.0f;
  }
  float creg[4] = {0.f, 0.f, 0.f, 0.f};

  for (int t = 0; t < T; ++t) {
    __syncthreads();  // h / gL from prev step settled
    const int t_in = in_rev ? (T - 1 - t) : t;
    const int t_out = out_rev ? (T - 1 - t) : t;

    // phase0: cooperative stage of xW[t] tile -> gL (16 rows x 1024 gates)
    {
      const int r0 = tid >> 6;           // 0..15
      const int cq = (tid & 63) << 4;    // 0,16,..,1008
      const int grow = row0 + r0;
      if (grow < nRows) {
        const _Float16* p = &xW[((size_t)grow * T + t_in) * 1024 + cq];
#if USE_ASYNC_LDS
        __builtin_amdgcn_global_load_async_to_lds_b128(
            (v4i*)p, (lds_v4i*)&gL[r0][cq], 0, 0);
        __builtin_amdgcn_global_load_async_to_lds_b128(
            (v4i*)(p + 8), (lds_v4i*)&gL[r0][cq + 8], 0, 0);
#else
        v8h x0 = *(const v8h*)p;
        v8h x1 = *(const v8h*)(p + 8);
        *(v8h*)&gL[r0][cq] = x0;
        *(v8h*)&gL[r0][cq + 8] = x1;
#endif
      }
    }

    v8f acc0 = v8f_zero(), acc1 = v8f_zero();
#pragma unroll
    for (int kk = 0; kk < 8; ++kk) {  // K = 256 in 32-chunks
      const int kb = kk * 32;
      v8h a0 = *(const v8h*)&hL[l16][kb + hsel * 8];
      v8h a1 = *(const v8h*)&hL[l16][kb + hsel * 8 + 16];
      v16h af = CAT16(a0, a1);
      const _Float16* bp0 = &WhhT[(size_t)(ncol0 + l16) * 256 + kb + hsel * 8];
      const _Float16* bp1 = &WhhT[(size_t)(ncol0 + 16 + l16) * 256 + kb + hsel * 8];
      if (kk < 7) {
        __builtin_prefetch(bp0 + 32, 0, 0);  // global_prefetch next K chunk
        __builtin_prefetch(bp1 + 32, 0, 0);
      }
      v8h b00 = *(const v8h*)bp0;
      v8h b01 = *(const v8h*)(bp0 + 16);
      acc0 = WMMA_F16(af, CAT16(b00, b01), acc0);
      v8h b10 = *(const v8h*)bp1;
      v8h b11 = *(const v8h*)(bp1 + 16);
      acc1 = WMMA_F16(af, CAT16(b10, b11), acc1);
    }

#if USE_ASYNC_LDS
    async_wait_all();
#endif
    __syncthreads();  // phase0 visible to all waves

    // phase3: gL += WMMA accumulators (C layout: VGPR v -> M=v (+8 hi lanes))
    {
      const int c0 = ncol0 + l16;
      const int c1 = ncol0 + 16 + l16;
#pragma unroll
      for (int v = 0; v < 8; ++v) {
        int m = v + hsel * 8;
        gL[m][c0] = (_Float16)((float)gL[m][c0] + acc0[v]);
        gL[m][c1] = (_Float16)((float)gL[m][c1] + acc1[v]);
      }
    }
    __syncthreads();

    // phase4: gate nonlinearity + state update; 4 (n,r) cells per thread
#pragma unroll
    for (int j = 0; j < 4; ++j) {
      int idx = tid + 1024 * j;
      int m = idx >> 8, r = idx & 255;
      int grow = row0 + m;
      float gi = (float)gL[m][r];
      float gf = (float)gL[m][256 + r];
      float gg = (float)gL[m][512 + r];
      float go = (float)gL[m][768 + r];
      float c = sigmf_(gf) * creg[j] + sigmf_(gi) * tanhf(gg);
      float h = sigmf_(go) * tanhf(c);
      if (grow >= nRows) { c = 0.0f; h = 0.0f; }
      creg[j] = c;
      hL[m][r] = (_Float16)h;
      if (grow < nRows) {
        size_t o = ((size_t)grow * T + t_out) * 256 + r;
        hs[o] = h;
        if (cs) cs[o] = c;
      }
    }
  }
}

// ---------------------------------------------------------------------------
// Small elementwise / attention kernels
// ---------------------------------------------------------------------------
__global__ void vec_add2(const float* a, const float* b, float* o, int n) {
  int i = blockIdx.x * blockDim.x + threadIdx.x;
  if (i < n) o[i] = a[i] + b[i];
}

__global__ void conv_f32_f16(const float* a, _Float16* o, int n) {
  int i = blockIdx.x * blockDim.x + threadIdx.x;
  if (i < n) o[i] = (_Float16)a[i];
}

// H = [hs_f|hs_b]*mask ; m = [cs_f|cs_b]*mask ; Hprev = H shifted by one step.
__global__ void build_H_m_Hprev(const float* hs_f, const float* cs_f,
                                const float* hs_b, const float* cs_b,
                                const int* len_ctx, float* H, float* mM, float* Hp) {
  size_t idx = blockIdx.x * (size_t)blockDim.x + threadIdx.x;
  if (idx >= (size_t)Bc * Tc * 2 * Rc) return;
  int d = (int)(idx % (2 * Rc));
  size_t bt = idx / (2 * Rc);
  int t = (int)(bt % Tc);
  int b = (int)(bt / Tc);
  int L = len_ctx[b];
  float msk = (t < L) ? 1.0f : 0.0f;
  size_t o256 = bt * (size_t)Rc;
  float hv, cv;
  if (d < Rc) { hv = hs_f[o256 + d];       cv = cs_f[o256 + d]; }
  else        { hv = hs_b[o256 + d - Rc];  cv = cs_b[o256 + d - Rc]; }
  H[idx] = hv * msk;
  mM[idx] = cv * msk;
  float hp = 0.0f;
  if (t > 0) {
    float mp = ((t - 1) < L) ? 1.0f : 0.0f;
    size_t o2 = (bt - 1) * (size_t)Rc;
    hp = ((d < Rc) ? hs_f[o2 + d] : hs_b[o2 + d - Rc]) * mp;
  }
  Hp[idx] = hp;
}

__global__ void extract_U(const float* sh_f, const float* sh_b, float* U) {
  int idx = blockIdx.x * blockDim.x + threadIdx.x;
  if (idx >= Bc * Sc * 2 * Rc) return;
  int d = idx % (2 * Rc);
  int row = idx / (2 * Rc);
  size_t last = ((size_t)row * Jc + (Jc - 1)) * Rc;
  U[idx] = (d < Rc) ? sh_f[last + d] : sh_b[last + d - Rc];
}

__global__ void sentinel_s(const float* eWh, const float* mM, float* sbuf, int n) {
  int i = blockIdx.x * blockDim.x + threadIdx.x;
  if (i < n) sbuf[i] = sigmf_(eWh[i]) * tanhf(mM[i]);
}

__global__ void attn_z(const float* cWh, const float* cWu, const float* Wv, float* z) {
  int idx = blockIdx.x * blockDim.x + threadIdx.x;
  if (idx >= Bc * Tc * Sc) return;
  int sp = idx & (Sc - 1);
  int bt = idx >> 5;
  int b = bt / Tc;
  const float* ch = cWh + (size_t)bt * MDUc;
  const float* cu = cWu + (size_t)(b * Sc + sp) * MDUc;
  float acc = 0.0f;
  for (int j = 0; j < MDUc; ++j) acc += tanhf(ch[j] + cu[j]) * Wv[j];
  z[idx] = acc;
}

__global__ void attn_zhat(const float* sWs, const float* cWh, const float* Wv, float* zh) {
  int bt = blockIdx.x * blockDim.x + threadIdx.x;
  if (bt >= Bc * Tc) return;
  const float* a = sWs + (size_t)bt * MDUc;
  const float* c = cWh + (size_t)bt * MDUc;
  float acc = 0.0f;
  for (int j = 0; j < MDUc; ++j) acc += tanhf(a[j] + c[j]) * Wv[j];
  zh[bt] = acc;
}

// Per (b,t): alpha=softmax(z), beta from softmax([z,zhat]); c_hat; assemble G.
__global__ __launch_bounds__(512) void attn_combine(
    const float* z, const float* zhat, const float* U,
    const float* sbuf, const float* H, float* G) {
  __shared__ float al[Sc];
  __shared__ float sh_beta;
  const int bt = blockIdx.x;
  const int b = bt >> 9;  // T = 512
  if (threadIdx.x == 0) {
    float zr[Sc + 1];
    float mx = -1e30f;
    for (int i = 0; i < Sc; ++i) { zr[i] = z[bt * Sc + i]; mx = fmaxf(mx, zr[i]); }
    zr[Sc] = zhat[bt];
    float mx33 = fmaxf(mx, zr[Sc]);
    float s32 = 0.0f, s33 = 0.0f;
    for (int i = 0; i < Sc; ++i) s32 += __expf(zr[i] - mx);
    for (int i = 0; i <= Sc; ++i) s33 += __expf(zr[i] - mx33);
    for (int i = 0; i < Sc; ++i) al[i] = __expf(zr[i] - mx) / s32;
    sh_beta = __expf(zr[Sc] - mx33) / s33;
  }
  __syncthreads();
  const float beta = sh_beta;
  const int d = threadIdx.x;  // 0..511
  float c = 0.0f;
  for (int sp = 0; sp < Sc; ++sp) c += al[sp] * U[((size_t)(b * Sc + sp)) * 512 + d];
  float ch = beta * sbuf[(size_t)bt * 512 + d] + (1.0f - beta) * c;
  G[(size_t)bt * 1024 + d] = H[(size_t)bt * 512 + d];
  G[(size_t)bt * 1024 + 512 + d] = ch;
}

// logit[bt,tag] = b_out + G.W_out[:1024] + [mh_f|mh_b].W_out[1024:1536]
__global__ void logits_kernel(const float* G, const float* mh_f, const float* mh_b,
                              const float* W_out, const float* b_out, float* out) {
  int idx = blockIdx.x * blockDim.x + threadIdx.x;
  if (idx >= Bc * Tc * TAGSc) return;
  int bt = idx / TAGSc, tag = idx % TAGSc;
  const float* w = W_out + (size_t)tag * (6 * Rc);
  const float* g = G + (size_t)bt * 1024;
  const float* mf = mh_f + (size_t)bt * Rc;
  const float* mb = mh_b + (size_t)bt * Rc;
  float acc = b_out[tag];
  for (int k = 0; k < 1024; ++k) acc += g[k] * w[k];
  for (int k = 0; k < Rc; ++k) acc += mf[k] * w[1024 + k];
  for (int k = 0; k < Rc; ++k) acc += mb[k] * w[1280 + k];
  out[idx] = acc;
}

// ---------------------------------------------------------------------------
extern "C" void kernel_launch(void* const* d_in, const int* in_sizes, int n_in,
                              void* d_out, int out_size, void* d_ws, size_t ws_size,
                              hipStream_t stream) {
  (void)in_sizes; (void)n_in; (void)out_size; (void)ws_size;

  const float* input_text = (const float*)d_in[0];
  const float* input_syn  = (const float*)d_in[1];
  const int*   len_ctx    = (const int*)d_in[3];
  const float *Wih[6], *Whh[6], *bih[6], *bhh[6];  // f1,b1,sf,sb,mf,mb
  for (int i = 0; i < 6; ++i) {
    Wih[i] = (const float*)d_in[5 + 4 * i];
    Whh[i] = (const float*)d_in[6 + 4 * i];
    bih[i] = (const float*)d_in[7 + 4 * i];
    bhh[i] = (const float*)d_in[8 + 4 * i];
  }
  const float* W_cWh = (const float*)d_in[29];
  const float* W_cWu = (const float*)d_in[30];
  const float* W_v   = (const float*)d_in[31];
  const float* W_sWh = (const float*)d_in[32];
  const float* W_sWu = (const float*)d_in[33];
  const float* W_Ws  = (const float*)d_in[34];
  const float* W_out = (const float*)d_in[35];
  const float* b_out = (const float*)d_in[36];

  char* wsb = (char*)d_ws;
  size_t off = 0;
  auto alloc = [&](size_t bytes) -> char* {
    char* p = wsb + off;
    off += (bytes + 255) & ~(size_t)255;
    return p;
  };

  const size_t BT = (size_t)Bc * Tc;       // 4096
  const size_t BSJ = (size_t)Bc * Sc * Jc; // 16384
  float* bias4[6];
  _Float16* WhhH[6];
  for (int i = 0; i < 6; ++i) bias4[i] = (float*)alloc(4096 * 4);
  for (int i = 0; i < 6; ++i) WhhH[i] = (_Float16*)alloc(1024 * 256 * 2);
  _Float16* xW_f1 = (_Float16*)alloc(BT * 1024 * 2);
  _Float16* xW_b1 = (_Float16*)alloc(BT * 1024 * 2);
  _Float16* xW_sf = (_Float16*)alloc(BSJ * 1024 * 2);
  _Float16* xW_sb = (_Float16*)alloc(BSJ * 1024 * 2);
  float* hs_f = (float*)alloc(BT * Rc * 4);
  float* cs_f = (float*)alloc(BT * Rc * 4);
  float* hs_b = (float*)alloc(BT * Rc * 4);
  float* cs_b = (float*)alloc(BT * Rc * 4);
  float* sh_f = (float*)alloc(BSJ * Rc * 4);
  float* sh_b = (float*)alloc(BSJ * Rc * 4);
  float* U    = (float*)alloc((size_t)Bc * Sc * 2 * Rc * 4);
  float* H    = (float*)alloc(BT * 2 * Rc * 4);
  float* mM   = (float*)alloc(BT * 2 * Rc * 4);
  float* Hp   = (float*)alloc(BT * 2 * Rc * 4);
  float* cWh  = (float*)alloc(BT * MDUc * 4);
  float* cWu  = (float*)alloc((size_t)Bc * Sc * MDUc * 4);
  float* Wsum = (float*)alloc((size_t)2 * Rc * 2 * Rc * 4);
  float* eWh  = (float*)alloc(BT * 2 * Rc * 4);
  float* sbuf = (float*)alloc(BT * 2 * Rc * 4);
  float* sWs  = (float*)alloc(BT * MDUc * 4);
  float* zb   = (float*)alloc(BT * Sc * 4);
  float* zh   = (float*)alloc(BT * 4);
  float* G    = (float*)alloc(BT * 4 * Rc * 4);
  _Float16* xW_mf = (_Float16*)alloc(BT * 1024 * 2);
  _Float16* xW_mb = (_Float16*)alloc(BT * 1024 * 2);
  float* mh_f = (float*)alloc(BT * Rc * 4);
  float* mh_b = (float*)alloc(BT * Rc * 4);

  auto cdiv = [](int a, int b) { return (a + b - 1) / b; };
  auto gemm = [&](const float* A, const float* Bw, const float* bias, void* C,
                  int M, int N, int K, int out16) {
    dim3 grid(cdiv(N, 128), cdiv(M, 64));
    gemm_wmma_nt<<<grid, 256, 0, stream>>>(A, Bw, bias, C, M, N, K, out16);
  };

  // 1) combined biases + f16 Whh copies
  for (int i = 0; i < 6; ++i) {
    vec_add2<<<cdiv(4 * Rc, 256), 256, 0, stream>>>(bih[i], bhh[i], bias4[i], 4 * Rc);
    conv_f32_f16<<<cdiv(1024 * 256, 256), 256, 0, stream>>>(Whh[i], WhhH[i], 1024 * 256);
  }

  // 2) input projections (WMMA GEMM, f16 out)
  gemm(input_text, Wih[0], bias4[0], xW_f1, (int)BT, 1024, Dc, 1);
  gemm(input_text, Wih[1], bias4[1], xW_b1, (int)BT, 1024, Dc, 1);
  gemm(input_syn,  Wih[2], bias4[2], xW_sf, (int)BSJ, 1024, Dc, 1);
  gemm(input_syn,  Wih[3], bias4[3], xW_sb, (int)BSJ, 1024, Dc, 1);

  // 3) recurrent scans (WMMA per step)
  lstm_scan_wmma<<<1, 1024, 0, stream>>>(xW_f1, WhhH[0], hs_f, cs_f, Bc, Tc, 0, 0);
  lstm_scan_wmma<<<1, 1024, 0, stream>>>(xW_b1, WhhH[1], hs_b, cs_b, Bc, Tc, 1, 0);
  lstm_scan_wmma<<<16, 1024, 0, stream>>>(xW_sf, WhhH[2], sh_f, nullptr, Bc * Sc, Jc, 0, 0);
  lstm_scan_wmma<<<16, 1024, 0, stream>>>(xW_sb, WhhH[3], sh_b, nullptr, Bc * Sc, Jc, 1, 0);

  // 4) U, H, m, Hprev
  extract_U<<<cdiv(Bc * Sc * 2 * Rc, 256), 256, 0, stream>>>(sh_f, sh_b, U);
  build_H_m_Hprev<<<cdiv((int)(BT * 2 * Rc), 256), 256, 0, stream>>>(
      hs_f, cs_f, hs_b, cs_b, len_ctx, H, mM, Hp);

  // 5) attention projections
  gemm(H, W_cWh, nullptr, cWh, (int)BT, MDUc, 2 * Rc, 0);
  gemm(U, W_cWu, nullptr, cWu, Bc * Sc, MDUc, 2 * Rc, 0);
  vec_add2<<<cdiv(2 * Rc * 2 * Rc, 256), 256, 0, stream>>>(W_sWh, W_sWu, Wsum, 2 * Rc * 2 * Rc);
  gemm(Hp, Wsum, nullptr, eWh, (int)BT, 2 * Rc, 2 * Rc, 0);
  sentinel_s<<<cdiv((int)(BT * 2 * Rc), 256), 256, 0, stream>>>(eWh, mM, sbuf, (int)(BT * 2 * Rc));
  gemm(sbuf, W_Ws, nullptr, sWs, (int)BT, MDUc, 2 * Rc, 0);

  // 6) scores, softmax, c_hat, G
  attn_z<<<cdiv((int)(BT * Sc), 256), 256, 0, stream>>>(cWh, cWu, W_v, zb);
  attn_zhat<<<cdiv((int)BT, 256), 256, 0, stream>>>(sWs, cWh, W_v, zh);
  attn_combine<<<(int)BT, 512, 0, stream>>>(zb, zh, U, sbuf, H, G);

  // 7) output BiLSTM over G
  gemm(G, Wih[4], bias4[4], xW_mf, (int)BT, 1024, 4 * Rc, 1);
  gemm(G, Wih[5], bias4[5], xW_mb, (int)BT, 1024, 4 * Rc, 1);
  lstm_scan_wmma<<<1, 1024, 0, stream>>>(xW_mf, WhhH[4], mh_f, nullptr, Bc, Tc, 0, 0);
  lstm_scan_wmma<<<1, 1024, 0, stream>>>(xW_mb, WhhH[5], mh_b, nullptr, Bc, Tc, 1, 1);

  // 8) logits
  logits_kernel<<<cdiv((int)(BT * TAGSc), 256), 256, 0, stream>>>(
      G, mh_f, mh_b, W_out, b_out, (float*)d_out);
}
